// AGFAttention_31490700214659
// MI455X (gfx1250) — matrix operand order repaired and checked by reading, not compiled, not run on hardware
//
#include <hip/hip_runtime.h>
#include <hip/hip_bf16.h>
#include <math.h>

// Problem constants (match reference)
#define B_     2
#define N_     4096
#define D_     256
#define H_     8
#define DH_    64
#define INNER_ 512
#define BN_    (B_ * N_)   // 8192
#define BH_    (B_ * H_)   // 16

typedef __attribute__((ext_vector_type(16))) _Float16 v16h;
typedef __attribute__((ext_vector_type(8)))  _Float16 v8h;
typedef __attribute__((ext_vector_type(8)))  float    v8f;

// D = A(16x32 f16) * B(32x16 f16) + C(16x16 f32)  -> v_wmma_f32_16x16x32_f16
__device__ __forceinline__ v8f wmma16(v16h a, v16h b, v8f c) {
  return __builtin_amdgcn_wmma_f32_16x16x32_f16(
      /*neg_a=*/false, a, /*neg_b=*/false, b,
      /*c_mod=*/(short)0, c, /*reuse_a=*/false, /*reuse_b=*/false);
}

// A-matrix fragment (16x32, row-major tile, ld in halves):
// lane<16: row=lane, K={0..15}; lane>=16: row=lane-16, K={16..31}
__device__ __forceinline__ v16h load_a_frag(const _Float16* tile, int ld) {
  const int lane = threadIdx.x & 31;
  const _Float16* p = tile + (size_t)(lane & 15) * ld + ((lane >> 4) << 4);
  v8h lo = *(const v8h*)p;        // 16B
  v8h hi = *(const v8h*)(p + 8);  // 16B
  return __builtin_shufflevector(lo, hi, 0,1,2,3,4,5,6,7,8,9,10,11,12,13,14,15);
}

// B-matrix fragment (32x16) from a K-transposed tile (tileT[n][k], ld in halves):
// lane = column n (mod 16), lanes>=16 read K+16; 16 contiguous halves (32B)
__device__ __forceinline__ v16h load_b_frag(const _Float16* tileT, int ld) {
  const int lane = threadIdx.x & 31;
  const _Float16* p = tileT + (size_t)(lane & 15) * ld + ((lane >> 4) << 4);
  return *(const v16h*)p;
}

// ---------------------------------------------------------------------------
// Kernel 1: projections. which=0: Q (pre-scaled by 1/sqrt(DH)), which=1: K,
// which=2: V with fused per-head LayerNorm; writes V^T f16 and Y = c0*v f32.
// Block tile 64(M) x 64(cols), 4 waves, K-loop over D_=256.
// ---------------------------------------------------------------------------
__global__ __launch_bounds__(128) void k_proj(
    const float* __restrict__ x,  const float* __restrict__ Wq,
    const float* __restrict__ Wk, const float* __restrict__ Wv,
    const float* __restrict__ gamma, const float* __restrict__ beta,
    const float* __restrict__ coeffs,
    _Float16* __restrict__ Qh, _Float16* __restrict__ Kh,
    _Float16* __restrict__ Vt, float* __restrict__ Y)
{
  const int which = blockIdx.z;
  const float* W = (which == 0) ? Wq : (which == 1) ? Wk : Wv;
  const int m0 = blockIdx.x * 64;   // row in [0, BN_)
  const int c0 = blockIdx.y * 64;   // col in [0, INNER_), head-aligned
  const int tid = threadIdx.x, lane = tid & 31, wv = tid >> 5;

  __shared__ __align__(64) _Float16 ldsX [64 * 32];   // x tile  [r][k] f16
  __shared__ __align__(64) _Float16 ldsWT[64 * 32];   // W tile  [n][k] f16 (transposed)
  __shared__ __align__(64) float    ldsV [64 * 64];   // LN staging (which==2)
  __shared__ float red[256];

  v8f acc[4] = {};
  for (int k0 = 0; k0 < D_; k0 += 32) {
    { // stage x -> f16 LDS (64 rows x 32 k)
      int r = tid & 63, hh = tid >> 6;
      const float* src = x + (size_t)(m0 + r) * D_ + k0 + hh * 16;
      _Float16* dst = ldsX + r * 32 + hh * 16;
      #pragma unroll
      for (int j = 0; j < 16; j += 4) {
        float4 f = *(const float4*)(src + j);
        dst[j+0] = (_Float16)f.x; dst[j+1] = (_Float16)f.y;
        dst[j+2] = (_Float16)f.z; dst[j+3] = (_Float16)f.w;
      }
    }
    { // stage W^T -> f16 LDS (64 n x 32 k)
      int kk = tid & 31, qq = tid >> 5;
      const float* src = W + (size_t)(k0 + kk) * INNER_ + c0 + qq * 16;
      #pragma unroll
      for (int j = 0; j < 16; ++j)
        ldsWT[(qq * 16 + j) * 32 + kk] = (_Float16)src[j];
    }
    __syncthreads();
    v16h a = load_a_frag(ldsX + wv * 16 * 32, 32);
    #pragma unroll
    for (int t = 0; t < 4; ++t) {
      v16h b = load_b_frag(ldsWT + t * 16 * 32, 32);
      acc[t] = wmma16(a, b, acc[t]);
    }
    __syncthreads();
  }

  if (which < 2) {
    const float sc = (which == 0) ? 0.125f : 1.0f;  // 1/sqrt(64) folded into Q
    _Float16* dst = (which == 0) ? Qh : Kh;
    #pragma unroll
    for (int t = 0; t < 4; ++t)
      #pragma unroll
      for (int i = 0; i < 8; ++i) {
        int gr = m0 + wv * 16 + i + ((lane >> 4) << 3);
        int gc = c0 + t * 16 + (lane & 15);
        int bb = gr >> 12, nn = gr & (N_ - 1);
        int hh = gc >> 6,  dd = gc & 63;
        dst[((size_t)(bb * H_ + hh) * N_ + nn) * DH_ + dd] = (_Float16)(acc[t][i] * sc);
      }
  } else {
    // dump f32 tile to LDS, then per-row LayerNorm over DH_=64
    #pragma unroll
    for (int t = 0; t < 4; ++t)
      #pragma unroll
      for (int i = 0; i < 8; ++i) {
        int rr = wv * 16 + i + ((lane >> 4) << 3);
        int cc = t * 16 + (lane & 15);
        ldsV[rr * 64 + cc] = acc[t][i];
      }
    __syncthreads();
    int r = tid >> 1, half = tid & 1;
    float s = 0.f, s2 = 0.f;
    #pragma unroll
    for (int j = 0; j < 32; ++j) {
      float v = ldsV[r * 64 + half * 32 + j];
      s += v; s2 += v * v;
    }
    red[tid] = s; red[128 + tid] = s2;
    __syncthreads();
    float S   = red[(r << 1)] + red[(r << 1) + 1];
    float S2  = red[128 + (r << 1)] + red[128 + (r << 1) + 1];
    float mu  = S * (1.0f / 64.0f);
    float var = S2 * (1.0f / 64.0f) - mu * mu;
    float rstd = rsqrtf(var + 1e-5f);
    int gr = m0 + r, bb = gr >> 12, nn = gr & (N_ - 1), hh = c0 >> 6;
    float c0h = coeffs[hh * 4 + 0];
    size_t bh = (size_t)(bb * H_ + hh);
    #pragma unroll
    for (int j = 0; j < 32; ++j) {
      int d = half * 32 + j;
      float v  = ldsV[r * 64 + d];
      float vn = (v - mu) * rstd * gamma[d] + beta[d];
      Vt[(bh * DH_ + d) * N_ + nn] = (_Float16)vn;        // V^T [bh][d][n] f16
      Y [(bh * N_ + nn) * DH_ + d] = c0h * vn;            // y init = c0 * v
    }
  }
}

// ---------------------------------------------------------------------------
// Kernel 2: E = exp(Q K^T) (scale pre-folded into Q) + per-row sums.
// Block tile 64(queries) x 128(keys), 8 waves, K-loop over DH_=64.
// Q/K read straight from global as WMMA fragments (contiguous b128 pairs).
// ---------------------------------------------------------------------------
__global__ __launch_bounds__(256) void k_scores(
    const _Float16* __restrict__ Qh, const _Float16* __restrict__ Kh,
    _Float16* __restrict__ E, float* __restrict__ rowsum)
{
  const int bh = blockIdx.z;
  const int n0 = blockIdx.x * 64;    // query tile
  const int m0 = blockIdx.y * 128;   // key tile
  const int lane = threadIdx.x & 31, wv = threadIdx.x >> 5;
  const int ms = wv & 3, ng = wv >> 2;  // m-strip, key half
  const _Float16* Qb = Qh + (size_t)bh * N_ * DH_;
  const _Float16* Kb = Kh + (size_t)bh * N_ * DH_;

  v8f acc[4] = {};
  #pragma unroll
  for (int k0 = 0; k0 < DH_; k0 += 32) {
    v16h a = load_a_frag(Qb + (size_t)(n0 + ms * 16) * DH_ + k0, DH_);
    #pragma unroll
    for (int t = 0; t < 4; ++t) {
      v16h b = load_b_frag(Kb + (size_t)(m0 + ng * 64 + t * 16) * DH_ + k0, DH_);
      acc[t] = wmma16(a, b, acc[t]);
    }
  }

  _Float16* Eb = E + (size_t)bh * N_ * N_;
  float part[8];
  #pragma unroll
  for (int i = 0; i < 8; ++i) part[i] = 0.f;
  #pragma unroll
  for (int t = 0; t < 4; ++t)
    #pragma unroll
    for (int i = 0; i < 8; ++i) {
      float e = __expf(acc[t][i]);    // logits ~O(0.1): no max-subtract needed
      int r = n0 + ms * 16 + i + ((lane >> 4) << 3);
      int c = m0 + ng * 64 + t * 16 + (lane & 15);
      Eb[(size_t)r * N_ + c] = (_Float16)e;
      part[i] += e;
    }
  // reduce the 16 lanes holding the same row, then atomically add row sums
  #pragma unroll
  for (int off = 1; off < 16; off <<= 1)
    #pragma unroll
    for (int i = 0; i < 8; ++i)
      part[i] += __shfl_xor(part[i], off, 32);
  if ((lane & 15) == 0) {
    int rbase = n0 + ms * 16 + ((lane >> 4) << 3);
    #pragma unroll
    for (int i = 0; i < 8; ++i)
      atomicAdd(&rowsum[(size_t)bh * N_ + rbase + i], part[i]);
  }
}

// ---------------------------------------------------------------------------
// Kernel 3: t_out = diag(1/rowsum) * E @ t_in ; Y += c_k * t_out ;
// write t_out^T f16 for the next polynomial power. 64(M) x 64(dh), 8 waves,
// K-loop over N_=4096 keys. All operands straight from global.
// ---------------------------------------------------------------------------
__global__ __launch_bounds__(256) void k_apply(
    const _Float16* __restrict__ E, const float* __restrict__ rowsum,
    const _Float16* __restrict__ TinT, _Float16* __restrict__ ToutT,
    float* __restrict__ Y, const float* __restrict__ coeffs, int kidx)
{
  const int bh = blockIdx.y;
  const int n0 = blockIdx.x * 64;
  const int lane = threadIdx.x & 31, wv = threadIdx.x >> 5;
  const int ms = wv & 3, ng = wv >> 2;  // m-strip, dh half
  const _Float16* Eb = E    + (size_t)bh * N_ * N_;
  const _Float16* Tb = TinT + (size_t)bh * DH_ * N_;

  v8f acc[2] = {};
  for (int k0 = 0; k0 < N_; k0 += 32) {
    v16h a = load_a_frag(Eb + (size_t)(n0 + ms * 16) * N_ + k0, N_);
    #pragma unroll
    for (int t = 0; t < 2; ++t) {
      v16h b = load_b_frag(Tb + (size_t)(ng * 32 + t * 16) * N_ + k0, N_);
      acc[t] = wmma16(a, b, acc[t]);
    }
  }

  const int h = bh & (H_ - 1);
  const float ck = coeffs[h * 4 + kidx];
  #pragma unroll
  for (int t = 0; t < 2; ++t)
    #pragma unroll
    for (int i = 0; i < 8; ++i) {
      int r = n0 + ms * 16 + i + ((lane >> 4) << 3);
      int d = ng * 32 + t * 16 + (lane & 15);
      float val = acc[t][i] / rowsum[(size_t)bh * N_ + r];  // softmax normalize
      Y[((size_t)bh * N_ + r) * DH_ + d] += ck * val;
      ToutT[((size_t)bh * DH_ + d) * N_ + r] = (_Float16)val;
    }
}

// ---------------------------------------------------------------------------
// Kernel 4: out = merge_heads(Y) @ Wo.  [8192 x 512] @ [512 x 256], f32 out.
// ---------------------------------------------------------------------------
__global__ __launch_bounds__(128) void k_out(
    const float* __restrict__ Y, const float* __restrict__ Wo,
    float* __restrict__ out)
{
  const int m0 = blockIdx.x * 64;
  const int c0 = blockIdx.y * 64;
  const int tid = threadIdx.x, lane = tid & 31, wv = tid >> 5;
  __shared__ __align__(64) _Float16 ldsA [64 * 32];
  __shared__ __align__(64) _Float16 ldsWT[64 * 32];

  v8f acc[4] = {};
  for (int k0 = 0; k0 < INNER_; k0 += 32) {
    { // stage merged-Y tile -> f16 (gather head-major layout)
      int r = tid & 63, hh = tid >> 6;
      int gr = m0 + r;
      int cbase = k0 + hh * 16;  // 16-chunk never crosses a head boundary
      const float* src = Y + ((size_t)((gr >> 12) * H_ + (cbase >> 6)) * N_
                              + (gr & (N_ - 1))) * DH_ + (cbase & 63);
      _Float16* dst = ldsA + r * 32 + hh * 16;
      #pragma unroll
      for (int j = 0; j < 16; j += 4) {
        float4 f = *(const float4*)(src + j);
        dst[j+0] = (_Float16)f.x; dst[j+1] = (_Float16)f.y;
        dst[j+2] = (_Float16)f.z; dst[j+3] = (_Float16)f.w;
      }
    }
    { // stage Wo^T -> f16
      int kk = tid & 31, qq = tid >> 5;
      const float* src = Wo + (size_t)(k0 + kk) * D_ + c0 + qq * 16;
      #pragma unroll
      for (int j = 0; j < 16; ++j)
        ldsWT[(qq * 16 + j) * 32 + kk] = (_Float16)src[j];
    }
    __syncthreads();
    v16h a = load_a_frag(ldsA + wv * 16 * 32, 32);
    #pragma unroll
    for (int t = 0; t < 4; ++t) {
      v16h b = load_b_frag(ldsWT + t * 16 * 32, 32);
      acc[t] = wmma16(a, b, acc[t]);
    }
    __syncthreads();
  }
  #pragma unroll
  for (int t = 0; t < 4; ++t)
    #pragma unroll
    for (int i = 0; i < 8; ++i) {
      int gr = m0 + wv * 16 + i + ((lane >> 4) << 3);
      int gc = c0 + t * 16 + (lane & 15);
      out[(size_t)gr * D_ + gc] = acc[t][i];
    }
}

// ---------------------------------------------------------------------------
// Workspace map (bytes):
//   Qh  f16 [BH][N][DH]   8 MB      Kh  f16            8 MB
//   Vt  f16 [BH][DH][N]   8 MB      Tt0/Tt1 f16        8 MB each
//   Y   f32 [BH][N][DH]  16 MB      rowsum f32 [BH][N] 256 KB
//   E   f16 [BH][N][N]  512 MB      total ~ 596 MB
// ---------------------------------------------------------------------------
extern "C" void kernel_launch(void* const* d_in, const int* in_sizes, int n_in,
                              void* d_out, int out_size, void* d_ws, size_t ws_size,
                              hipStream_t stream) {
  (void)in_sizes; (void)n_in; (void)out_size; (void)ws_size;
  const float* x      = (const float*)d_in[0];
  const float* Wq     = (const float*)d_in[1];
  const float* Wk     = (const float*)d_in[2];
  const float* Wv     = (const float*)d_in[3];
  const float* Wo     = (const float*)d_in[4];
  const float* gamma  = (const float*)d_in[5];
  const float* beta   = (const float*)d_in[6];
  const float* coeffs = (const float*)d_in[7];
  float* out = (float*)d_out;

  char* ws = (char*)d_ws;
  const size_t szH = (size_t)BH_ * N_ * DH_ * sizeof(_Float16); // 8 MB
  _Float16* Qh  = (_Float16*)(ws + 0 * szH);
  _Float16* Kh  = (_Float16*)(ws + 1 * szH);
  _Float16* Vt  = (_Float16*)(ws + 2 * szH);
  _Float16* Tt0 = (_Float16*)(ws + 3 * szH);
  _Float16* Tt1 = (_Float16*)(ws + 4 * szH);
  float*    Y   = (float*)   (ws + 5 * szH);
  size_t off = 5 * szH + (size_t)BH_ * N_ * DH_ * sizeof(float);
  float*    rowsum = (float*)(ws + off);
  off += (size_t)BH_ * N_ * sizeof(float);
  _Float16* E   = (_Float16*)(ws + off);

  hipMemsetAsync(rowsum, 0, (size_t)BH_ * N_ * sizeof(float), stream);

  dim3 gp(BN_ / 64, INNER_ / 64, 3);
  k_proj<<<gp, 128, 0, stream>>>(x, Wq, Wk, Wv, gamma, beta, coeffs, Qh, Kh, Vt, Y);

  dim3 gs(N_ / 64, N_ / 128, BH_);
  k_scores<<<gs, 256, 0, stream>>>(Qh, Kh, E, rowsum);

  dim3 ga(N_ / 64, BH_);
  k_apply<<<ga, 256, 0, stream>>>(E, rowsum, Vt,  Tt0, Y, coeffs, 1);
  k_apply<<<ga, 256, 0, stream>>>(E, rowsum, Tt0, Tt1, Y, coeffs, 2);
  k_apply<<<ga, 256, 0, stream>>>(E, rowsum, Tt1, Tt0, Y, coeffs, 3);

  dim3 go(BN_ / 64, D_ / 64);
  k_out<<<go, 128, 0, stream>>>(Y, Wo, out);
}
